// QINCo_1322849927709
// MI455X (gfx1250) — matrix operand order repaired and checked by reading, not compiled, not run on hardware
//
#include <hip/hip_runtime.h>

#define DD 128
#define KK 256
#define MM 8
#define LL 2
#define HH 256
#define BB 1024
#define DDP (DD + 4)   // padded f32 row (bank-conflict-free: +4 words)
#define HHP (HH + 8)   // padded bf16 row (+8 halves = +4 words)

typedef __attribute__((ext_vector_type(16))) __bf16 v16bf;
typedef __attribute__((ext_vector_type(8)))  float  v8f;

// ---------------- WMMA fragment helpers (wave32; k = 16*(lane/16) + e in both A and B) ----------------

__device__ __forceinline__ v8f wmma_bf16(v16bf a, v16bf b, v8f c) {
  return __builtin_amdgcn_wmma_f32_16x16x32_bf16(false, a, false, b, (short)0, c,
                                                 false, false);
}

// A fragment: 16x32 bf16 from f32 source. Element (m,k) at src[m*sM + k].
// lanes 0-15: row M=lane, K=0..15; lanes 16-31: same M, K=16..31
__device__ __forceinline__ v16bf load_A_f32(const float* src, int sM, int lane) {
  int m = lane & 15, hi = (lane >> 4) & 1;
  v16bf a;
#pragma unroll
  for (int e = 0; e < 16; ++e) {
    int k = hi * 16 + e;
    a[e] = (__bf16)src[m * sM + k];
  }
  return a;
}

__device__ __forceinline__ v16bf load_A_bf16(const __bf16* src, int sM, int lane) {
  int m = lane & 15, hi = (lane >> 4) & 1;
  v16bf a;
#pragma unroll
  for (int e = 0; e < 16; ++e) {
    int k = hi * 16 + e;
    a[e] = src[m * sM + k];
  }
  return a;
}

// B fragment: 32x16 bf16. Element (k,n) at src[k*sK + n*sN].
// lane = column N (mod 16); lanes 0-15 hold K=0..15, lanes 16-31 hold K=16..31.
__device__ __forceinline__ v16bf load_B_bf16(const __bf16* src, long sK, long sN, int lane) {
  int n = lane & 15, hi = (lane >> 4) & 1;
  v16bf b;
#pragma unroll
  for (int e = 0; e < 16; ++e) {
    int k = hi * 16 + e;
    b[e] = src[(long)k * sK + (long)n * sN];
  }
  return b;
}

__device__ __forceinline__ v16bf load_B_f32(const float* src, long sK, long sN, int lane) {
  int n = lane & 15, hi = (lane >> 4) & 1;
  v16bf b;
#pragma unroll
  for (int e = 0; e < 16; ++e) {
    int k = hi * 16 + e;
    b[e] = (__bf16)src[(long)k * sK + (long)n * sN];
  }
  return b;
}

// ---------------- K1: f32 -> bf16 weight conversion (once) ----------------

__global__ void k_convert(const float* __restrict__ wc, const float* __restrict__ w1,
                          const float* __restrict__ w2,
                          __bf16* __restrict__ wcb, __bf16* __restrict__ w1b,
                          __bf16* __restrict__ w2b) {
  int i = blockIdx.x * blockDim.x + threadIdx.x;
  const int NWC = 7 * DD * 2 * DD;   // 229376
  const int NW  = 7 * LL * HH * DD;  // 458752
  if (i < NWC) wcb[i] = (__bf16)wc[i];
  if (i < NW) { w1b[i] = (__bf16)w1[i]; w2b[i] = (__bf16)w2[i]; }
}

// ---------------- K2: row norms ||x||^2 and ||c0||^2 ----------------

__global__ void k_norms(const float* __restrict__ x, const float* __restrict__ cb0,
                        float* __restrict__ xn, float* __restrict__ cn0) {
  int r = blockIdx.x * blockDim.x + threadIdx.x;
  if (r < BB) {
    float s = 0.f;
    for (int d = 0; d < DD; ++d) { float v = x[(long)r * DD + d]; s += v * v; }
    xn[r] = s;
  } else if (r < BB + KK) {
    int k = r - BB;
    float s = 0.f;
    for (int d = 0; d < DD; ++d) { float v = cb0[(long)k * DD + d]; s += v * v; }
    cn0[k] = s;
  }
}

// ---------------- K3: code0 = argmin_k ||x - c0_k||^2 via WMMA x@c0^T ----------------

__global__ void k_code0(const float* __restrict__ x, const float* __restrict__ cb0,
                        const float* __restrict__ xn, const float* __restrict__ cn0,
                        int* __restrict__ codes_i, float* __restrict__ codes_f,
                        float* __restrict__ side0) {
  __shared__ float sdist[16][KK];
  __shared__ int scode[16];
  int b0 = blockIdx.x * 16;
  int wave = threadIdx.x >> 5, lane = threadIdx.x & 31;
  // A fragments (x rows b0..b0+15) are shared by this wave's 2 column tiles: hoist.
  v16bf a0 = load_A_f32(x + (long)b0 * DD + 0,  DD, lane);
  v16bf a1 = load_A_f32(x + (long)b0 * DD + 32, DD, lane);
  v16bf a2 = load_A_f32(x + (long)b0 * DD + 64, DD, lane);
  v16bf a3 = load_A_f32(x + (long)b0 * DD + 96, DD, lane);
#pragma unroll
  for (int t = 0; t < 2; ++t) {
    int ct = wave * 2 + t;  // 16 col-tiles of 16 codes
    v8f acc = {};
    acc = wmma_bf16(a0, load_B_f32(cb0 + (long)(ct * 16) * DD + 0,  1, DD, lane), acc);
    acc = wmma_bf16(a1, load_B_f32(cb0 + (long)(ct * 16) * DD + 32, 1, DD, lane), acc);
    acc = wmma_bf16(a2, load_B_f32(cb0 + (long)(ct * 16) * DD + 64, 1, DD, lane), acc);
    acc = wmma_bf16(a3, load_B_f32(cb0 + (long)(ct * 16) * DD + 96, 1, DD, lane), acc);
    int n = lane & 15, hi = (lane >> 4) & 1;
#pragma unroll
    for (int i = 0; i < 8; ++i) sdist[i + hi * 8][ct * 16 + n] = -2.f * acc[i];
  }
  __syncthreads();
  if (threadIdx.x < 16) {
    int r = threadIdx.x, b = b0 + r;
    float best = 3.4e38f; int bi = 0;
    for (int k = 0; k < KK; ++k) {
      float dv = xn[b] + cn0[k] + sdist[r][k];
      if (dv < best) { best = dv; bi = k; }
    }
    scode[r] = bi;
    codes_i[b * MM + 0] = bi;
    codes_f[b * MM + 0] = (float)bi;
  }
  __syncthreads();
  for (int idx = threadIdx.x; idx < 16 * DD; idx += blockDim.x) {
    int r = idx >> 7, d = idx & (DD - 1);
    side0[(long)(b0 + r) * DD + d] = cb0[(long)scode[r] * DD + d];
  }
}

// ---------------- K4/K5: t1 = cb + cb@Wz^T + bc  /  t2 = xhat@Wx^T ----------------
// One row-tile of 16 per workgroup; 8 waves = 8 col-tiles of D=128.

__global__ void k_tprep(const float* __restrict__ A, const __bf16* __restrict__ wcb_m,
                        int wcoff, const float* __restrict__ bc_m,
                        float* __restrict__ outT, int addA) {
  int rt = blockIdx.x;
  int wave = threadIdx.x >> 5, lane = threadIdx.x & 31;
  int ct = wave;  // 0..7
  const float* Arow = A + (long)rt * 16 * DD;
  v8f acc = {};
#pragma unroll
  for (int kk = 0; kk < DD; kk += 32) {
    v16bf a  = load_A_f32(Arow + kk, DD, lane);
    // B(k,n) = Wc_m[(ct*16+n)*2D + wcoff + kk + k]
    v16bf bm = load_B_bf16(wcb_m + (long)(ct * 16) * (2 * DD) + wcoff + kk, 1, 2 * DD, lane);
    acc = wmma_bf16(a, bm, acc);
  }
  int n = lane & 15, hi = (lane >> 4) & 1;
#pragma unroll
  for (int i = 0; i < 8; ++i) {
    int r = rt * 16 + i + hi * 8, c = ct * 16 + n;
    float v = acc[i];
    if (addA) v += A[(long)r * DD + c] + bc_m[c];
    outT[(long)r * DD + c] = v;
  }
}

// ---------------- K6/K8: 2-layer residual MLP over 128 rows (the hot kernel) -------
// WINNER==0: rows = one batch element b x 128 candidate codes; emits distances.
// WINNER==1: rows = 128 batch elements with their winning code; emits xhat_next.
// 8 waves; wave w owns row-tile w (rows w*16..w*16+15) for both GEMMs, so its
// A fragments are loop-invariant registers while B streams from L2.

template <int WINNER>
__global__ void k_mlp(const float* __restrict__ t1, const float* __restrict__ t2,
                      const __bf16* __restrict__ w1b_m, const __bf16* __restrict__ w2b_m,
                      const float* __restrict__ x, const float* __restrict__ xhat,
                      const float* __restrict__ xn, const int* __restrict__ codes_i,
                      int mstep, float* __restrict__ dist, float* __restrict__ xhat_next) {
  __shared__ float  zs[128][DDP];   // 67.5 KB, bank-conflict-free rows
  __shared__ __bf16 hs[128][HHP];   // 67.5 KB
  __shared__ float  xb[DD], xh[DD];
  __shared__ float  partial[128][2];

  int wave = threadIdx.x >> 5, lane = threadIdx.x & 31;
  int b = 0, k0 = 0, b0 = 0;

  if (!WINNER) {
    b  = blockIdx.x >> 1;
    k0 = (blockIdx.x & 1) * 128;
    for (int idx = threadIdx.x; idx < 128 * DD; idx += blockDim.x) {
      int r = idx >> 7, d = idx & (DD - 1);
      zs[r][d] = t1[(long)(k0 + r) * DD + d] + t2[(long)b * DD + d];
    }
    for (int d = threadIdx.x; d < DD; d += blockDim.x) {
      xb[d] = x[(long)b * DD + d];
      xh[d] = xhat[(long)b * DD + d];
    }
  } else {
    b0 = blockIdx.x * 128;
    for (int idx = threadIdx.x; idx < 128 * DD; idx += blockDim.x) {
      int r = idx >> 7, d = idx & (DD - 1);
      int bb = b0 + r;
      int kw = codes_i[bb * MM + mstep + 1];
      zs[r][d] = t1[(long)kw * DD + d] + t2[(long)bb * DD + d];
    }
  }
  __syncthreads();

  for (int l = 0; l < LL; ++l) {
    const __bf16* W1l = w1b_m + (long)l * HH * DD;
    const __bf16* W2l = w2b_m + (long)l * DD * HH;
    __builtin_prefetch(W1l, 0, 3);
    __builtin_prefetch(W2l, 0, 3);

    // GEMM1: h = relu(z @ W1^T)  [128x128]x[128x256]; wave row-tile fixed = wave.
    {
      v16bf a0 = load_A_f32(&zs[wave * 16][0],  DDP, lane);
      v16bf a1 = load_A_f32(&zs[wave * 16][32], DDP, lane);
      v16bf a2 = load_A_f32(&zs[wave * 16][64], DDP, lane);
      v16bf a3 = load_A_f32(&zs[wave * 16][96], DDP, lane);
#pragma unroll 1
      for (int ct = 0; ct < 16; ++ct) {
        const __bf16* Bc = W1l + (long)(ct * 16) * DD;
        v8f acc = {};
        acc = wmma_bf16(a0, load_B_bf16(Bc + 0,  1, DD, lane), acc);
        acc = wmma_bf16(a1, load_B_bf16(Bc + 32, 1, DD, lane), acc);
        acc = wmma_bf16(a2, load_B_bf16(Bc + 64, 1, DD, lane), acc);
        acc = wmma_bf16(a3, load_B_bf16(Bc + 96, 1, DD, lane), acc);
        int n = lane & 15, hi = (lane >> 4) & 1;
#pragma unroll
        for (int i = 0; i < 8; ++i) {
          float v = acc[i];
          v = v > 0.f ? v : 0.f;
          hs[wave * 16 + i + hi * 8][ct * 16 + n] = (__bf16)v;
        }
      }
    }
    __syncthreads();

    // GEMM2: z += h @ W2^T  [128x256]x[256x128]; wave row-tile fixed = wave.
    {
      v16bf ha[8];
#pragma unroll
      for (int q = 0; q < 8; ++q) ha[q] = load_A_bf16(&hs[wave * 16][q * 32], HHP, lane);
#pragma unroll 1
      for (int ct = 0; ct < 8; ++ct) {
        const __bf16* Bc = W2l + (long)(ct * 16) * HH;
        v8f acc = {};
#pragma unroll
        for (int q = 0; q < 8; ++q)
          acc = wmma_bf16(ha[q], load_B_bf16(Bc + q * 32, 1, HH, lane), acc);
        int n = lane & 15, hi = (lane >> 4) & 1;
#pragma unroll
        for (int i = 0; i < 8; ++i) {
          zs[wave * 16 + i + hi * 8][ct * 16 + n] += acc[i];
        }
      }
    }
    __syncthreads();
  }

  if (!WINNER) {
    // dist[b,k] = xn[b] + sum_d (z+xh)^2 - 2 x_d (z+xh); 2 threads per row,
    // cyclic start so the 64-iteration scans spread over LDS banks.
    int r = threadIdx.x >> 1, half = threadIdx.x & 1;
    float s = 0.f;
    for (int i = 0; i < 64; ++i) {
      int d = half * 64 + ((i + r) & 63);
      float zf = zs[r][d] + xh[d];
      s += zf * zf - 2.f * xb[d] * zf;
    }
    partial[r][half] = s;
    __syncthreads();
    if (threadIdx.x < 128) {
      int rr = threadIdx.x;
      dist[(long)b * KK + k0 + rr] = xn[b] + partial[rr][0] + partial[rr][1];
    }
  } else {
    for (int idx = threadIdx.x; idx < 128 * DD; idx += blockDim.x) {
      int r = idx >> 7, d = idx & (DD - 1);
      int bb = b0 + r;
      xhat_next[(long)bb * DD + d] = zs[r][d] + xhat[(long)bb * DD + d];
    }
  }
}

// ---------------- K7: argmin over K=256 distances per batch row ----------------

__global__ void k_argmin(const float* __restrict__ dist, int* __restrict__ codes_i,
                         float* __restrict__ codes_f, int mstep) {
  int b = blockIdx.x * blockDim.x + threadIdx.x;
  if (b >= BB) return;
  float best = 3.4e38f; int bi = 0;
  for (int k = 0; k < KK; ++k) {
    float v = dist[(long)b * KK + k];
    if (v < best) { best = v; bi = k; }
  }
  codes_i[b * MM + mstep + 1] = bi;
  codes_f[b * MM + mstep + 1] = (float)bi;
}

// ---------------- K9: final xhat = side[M-1] ----------------

__global__ void k_copy(const float* __restrict__ src, float* __restrict__ dst, int n) {
  int i = blockIdx.x * blockDim.x + threadIdx.x;
  if (i < n) dst[i] = src[i];
}

// ---------------- launch ----------------

extern "C" void kernel_launch(void* const* d_in, const int* in_sizes, int n_in,
                              void* d_out, int out_size, void* d_ws, size_t ws_size,
                              hipStream_t stream) {
  const float* x   = (const float*)d_in[0];  // [BB,DD]
  const float* cb0 = (const float*)d_in[1];  // [KK,DD]
  const float* cbs = (const float*)d_in[2];  // [7,KK,DD]
  const float* Wc  = (const float*)d_in[3];  // [7,DD,2DD]
  const float* bc  = (const float*)d_in[4];  // [7,DD]
  const float* W1  = (const float*)d_in[5];  // [7,LL,HH,DD]
  const float* W2  = (const float*)d_in[6];  // [7,LL,DD,HH]

  float* out       = (float*)d_out;
  float* out_xhat  = out;                       // BB*DD
  float* out_codes = out + (long)BB * DD;       // BB*MM
  float* out_side  = out_codes + (long)BB * MM; // MM*BB*DD

  char* w = (char*)d_ws;
  __bf16* wcb = (__bf16*)w; w += sizeof(__bf16) * 7 * DD * 2 * DD;
  __bf16* w1b = (__bf16*)w; w += sizeof(__bf16) * 7 * LL * HH * DD;
  __bf16* w2b = (__bf16*)w; w += sizeof(__bf16) * 7 * LL * DD * HH;
  float* t1   = (float*)w;  w += sizeof(float) * KK * DD;
  float* t2   = (float*)w;  w += sizeof(float) * BB * DD;
  float* xn   = (float*)w;  w += sizeof(float) * BB;
  float* cn0  = (float*)w;  w += sizeof(float) * KK;
  float* dist = (float*)w;  w += sizeof(float) * BB * KK;
  int* codes_i = (int*)w;   w += sizeof(int) * BB * MM;

  k_convert<<<1792, 256, 0, stream>>>(Wc, W1, W2, wcb, w1b, w2b);
  k_norms<<<(BB + KK + 255) / 256, 256, 0, stream>>>(x, cb0, xn, cn0);
  k_code0<<<BB / 16, 256, 0, stream>>>(x, cb0, xn, cn0, codes_i, out_codes, out_side);

  for (int m = 0; m < MM - 1; ++m) {
    const float* xhat_m = out_side + (long)m * BB * DD;        // side[m]
    float*       xhat_n = out_side + (long)(m + 1) * BB * DD;  // side[m+1]
    const __bf16* wcb_m = wcb + (long)m * DD * 2 * DD;
    const __bf16* w1b_m = w1b + (long)m * LL * HH * DD;
    const __bf16* w2b_m = w2b + (long)m * LL * DD * HH;

    // t1[k,:] = cb[k,:] + cb@Wz^T + bc ;  t2[b,:] = xhat@Wx^T
    k_tprep<<<KK / 16, 256, 0, stream>>>(cbs + (long)m * KK * DD, wcb_m, 0,
                                         bc + (long)m * DD, t1, 1);
    k_tprep<<<BB / 16, 256, 0, stream>>>(xhat_m, wcb_m, DD,
                                         bc + (long)m * DD, t2, 0);
    // candidate MLP + distances (dominant kernel)
    k_mlp<0><<<BB * 2, 256, 0, stream>>>(t1, t2, w1b_m, w2b_m, x, xhat_m, xn,
                                         codes_i, m, dist, nullptr);
    k_argmin<<<(BB + 255) / 256, 256, 0, stream>>>(dist, codes_i, out_codes, m);
    // recompute winners only -> side[m+1]
    k_mlp<1><<<BB / 128, 256, 0, stream>>>(t1, t2, w1b_m, w2b_m, x, xhat_m, xn,
                                           codes_i, m, dist, xhat_n);
  }

  k_copy<<<(BB * DD + 255) / 256, 256, 0, stream>>>(out_side + (long)(MM - 1) * BB * DD,
                                                    out_xhat, BB * DD);
}
